// MoE_68676527063394
// MI455X (gfx1250) — matrix operand order repaired and checked by reading, not compiled, not run on hardware
//
#include <hip/hip_runtime.h>

// MoE top-2 SwiGLU, dense-masked formulation (identical to reference):
//   scale[t,e] = softmax(x@gate_w)[t,e] if e in top2(t) else 0
//   h[t, e*H+c] = silu((scale*x) @ w1_e) * ((scale*x) @ w3_e)   (bf16, ws)
//   out = h[T, E*H] @ w2_flat[E*H, D]                            (fp32)
// GEMMs run on v_wmma_f32_16x16x32_bf16 (fp32 accumulate).
// Block tile 128(M) x 64(N); 4 waves, each wave owns 32 rows x 64 cols
// (2 A-fragments reused across 4 N-subtiles -> 16 WMMAs / 20 ds_load_b128).

#define T_TOK 1024
#define DIM   1024
#define HID   512
#define NEXP  8
#define KTOT  (NEXP * HID)   // 4096
#define LDSS  40             // 32 bf16 + 8 pad ushorts, keeps 16B alignment

typedef __attribute__((ext_vector_type(16))) __bf16        v16bf;
typedef __attribute__((ext_vector_type(8)))  float         v8f;
typedef __attribute__((ext_vector_type(4)))  unsigned int  u32x4;
typedef __attribute__((ext_vector_type(4)))  float         f32x4;

union Frag {
  v16bf v;
  u32x4 q[2];
};

__device__ __forceinline__ unsigned short f2bf(float f) {
  unsigned int u = __float_as_uint(f);
  unsigned int r = (u + 0x7FFFu + ((u >> 16) & 1u)) >> 16;  // RNE
  return (unsigned short)r;
}
__device__ __forceinline__ unsigned int pk2(float a, float b) {
  return (unsigned int)f2bf(a) | ((unsigned int)f2bf(b) << 16);
}

// ---------------------------------------------------------------- routing
__global__ void moe_route_kernel(const float* __restrict__ x,
                                 const float* __restrict__ gw,
                                 float* __restrict__ scale) {
  int t = blockIdx.x * blockDim.x + threadIdx.x;
  if (t >= T_TOK) return;
  const float* xr = x + (size_t)t * DIM;
  float l[NEXP];
#pragma unroll
  for (int e = 0; e < NEXP; ++e) l[e] = 0.f;
  for (int d = 0; d < DIM; ++d) {
    float xv = xr[d];
    const float* g = gw + (size_t)d * NEXP;
#pragma unroll
    for (int e = 0; e < NEXP; ++e) l[e] += xv * g[e];
  }
  float mx = l[0];
#pragma unroll
  for (int e = 1; e < NEXP; ++e) mx = fmaxf(mx, l[e]);
  float p[NEXP], sum = 0.f;
#pragma unroll
  for (int e = 0; e < NEXP; ++e) { p[e] = __expf(l[e] - mx); sum += p[e]; }
  float inv = __builtin_amdgcn_rcpf(sum);
#pragma unroll
  for (int e = 0; e < NEXP; ++e) p[e] *= inv;
  int i0 = 0;
#pragma unroll
  for (int e = 1; e < NEXP; ++e) if (p[e] > p[i0]) i0 = e;
  int i1 = (i0 == 0) ? 1 : 0;
#pragma unroll
  for (int e = 0; e < NEXP; ++e) if (e != i0 && p[e] > p[i1]) i1 = e;
#pragma unroll
  for (int e = 0; e < NEXP; ++e)
    scale[(size_t)t * NEXP + e] = (e == i0 || e == i1) ? p[e] : 0.f;
}

// A-fragment per ISA table: lane = 16*hi + m; VGPR0-3 hold K = hi?8..15:0..7,
// VGPR4-7 hold K = 16 + same  -> two 16B LDS reads at byte offsets hi*16, 32+hi*16.
// B stored in LDS as [n][k]; lane reads K as for A (hi?8..15:0..7, then 16 + same) of column n -> two 16B reads.

// ---------------------------------------------------------------- pass 1
__global__ void __launch_bounds__(128)
moe_ffn1_kernel(const float* __restrict__ x,
                const float* __restrict__ scale,
                const float* __restrict__ w1,
                const float* __restrict__ w3,
                unsigned short* __restrict__ hbuf) {
  __shared__ __align__(16) unsigned short At[128 * LDSS];  // 10 KB
  __shared__ __align__(16) unsigned short Bg[64 * LDSS];   // 5 KB
  __shared__ __align__(16) unsigned short Bu[64 * LDSS];   // 5 KB

  const int tid  = threadIdx.x;
  const int lane = tid & 31;
  const int wid  = tid >> 5;
  const int e    = blockIdx.z;
  const int tM0  = blockIdx.x * 128;
  const int tN0  = blockIdx.y * 64;
  const int m16  = lane & 15;
  const int hi   = lane >> 4;

  const v8f vz = {0.f, 0.f, 0.f, 0.f, 0.f, 0.f, 0.f, 0.f};
  v8f accg[2][4], accu[2][4];
#pragma unroll
  for (int mi = 0; mi < 2; ++mi)
#pragma unroll
    for (int s = 0; s < 4; ++s) { accg[mi][s] = vz; accu[mi][s] = vz; }

  // A loader: one thread per row (128 rows x 32 k fp32 -> bf16*scale)
  const float sc = scale[(size_t)(tM0 + tid) * NEXP + e];
  const float* xrow = x + (size_t)(tM0 + tid) * DIM;
  u32x4* aout = (u32x4*)&At[tid * LDSS];

  for (int k0 = 0; k0 < DIM; k0 += 32) {
    f32x4 f[8];
#pragma unroll
    for (int j = 0; j < 8; ++j) f[j] = *(const f32x4*)(xrow + k0 + 4 * j);
#pragma unroll
    for (int j = 0; j < 4; ++j) {
      u32x4 u;
      u.x = pk2(f[2 * j].x * sc, f[2 * j].y * sc);
      u.y = pk2(f[2 * j].z * sc, f[2 * j].w * sc);
      u.z = pk2(f[2 * j + 1].x * sc, f[2 * j + 1].y * sc);
      u.w = pk2(f[2 * j + 1].z * sc, f[2 * j + 1].w * sc);
      aout[j] = u;
    }

    // B tiles (w1 gate, w3 up), stored transposed [n][k]
#pragma unroll
    for (int i = 0; i < 4; ++i) {
      int idx = tid + 128 * i;            // 0..511
      int k = idx >> 4;                   // 0..31
      int n = (idx & 15) * 4;             // 0..60
      size_t go = ((size_t)e * DIM + (k0 + k)) * HID + tN0 + n;
      f32x4 wg = *(const f32x4*)(w1 + go);
      f32x4 wu = *(const f32x4*)(w3 + go);
      Bg[(n + 0) * LDSS + k] = f2bf(wg.x);
      Bg[(n + 1) * LDSS + k] = f2bf(wg.y);
      Bg[(n + 2) * LDSS + k] = f2bf(wg.z);
      Bg[(n + 3) * LDSS + k] = f2bf(wg.w);
      Bu[(n + 0) * LDSS + k] = f2bf(wu.x);
      Bu[(n + 1) * LDSS + k] = f2bf(wu.y);
      Bu[(n + 2) * LDSS + k] = f2bf(wu.z);
      Bu[(n + 3) * LDSS + k] = f2bf(wu.w);
    }
    __syncthreads();

    Frag fa0, fa1;
    const u32x4* ar0 = (const u32x4*)&At[(wid * 32 + m16) * LDSS];
    fa0.q[0] = ar0[hi];
    fa0.q[1] = ar0[2 + hi];
    const u32x4* ar1 = (const u32x4*)&At[(wid * 32 + 16 + m16) * LDSS];
    fa1.q[0] = ar1[hi];
    fa1.q[1] = ar1[2 + hi];
#pragma unroll
    for (int s = 0; s < 4; ++s) {
      Frag fb;
      const u32x4* br = (const u32x4*)&Bg[(s * 16 + m16) * LDSS];
      fb.q[0] = br[hi]; fb.q[1] = br[2 + hi];
      accg[0][s] = __builtin_amdgcn_wmma_f32_16x16x32_bf16(
          false, fa0.v, false, fb.v, (short)0, accg[0][s], false, false);
      accg[1][s] = __builtin_amdgcn_wmma_f32_16x16x32_bf16(
          false, fa1.v, false, fb.v, (short)0, accg[1][s], false, false);
      const u32x4* ur = (const u32x4*)&Bu[(s * 16 + m16) * LDSS];
      fb.q[0] = ur[hi]; fb.q[1] = ur[2 + hi];
      accu[0][s] = __builtin_amdgcn_wmma_f32_16x16x32_bf16(
          false, fa0.v, false, fb.v, (short)0, accu[0][s], false, false);
      accu[1][s] = __builtin_amdgcn_wmma_f32_16x16x32_bf16(
          false, fa1.v, false, fb.v, (short)0, accu[1][s], false, false);
    }
    __syncthreads();
  }

  // epilogue: h = silu(gate) * up -> bf16 at hbuf[t][e*HID + c]
#pragma unroll
  for (int mi = 0; mi < 2; ++mi) {
#pragma unroll
    for (int s = 0; s < 4; ++s) {
#pragma unroll
      for (int r = 0; r < 8; ++r) {
        int t = tM0 + wid * 32 + mi * 16 + hi * 8 + r;
        int c = tN0 + s * 16 + m16;
        float g = accg[mi][s][r];
        float u = accu[mi][s][r];
        float h = g * __builtin_amdgcn_rcpf(1.f + __expf(-g)) * u;
        hbuf[(size_t)t * KTOT + e * HID + c] = f2bf(h);
      }
    }
  }
}

// ---------------------------------------------------------------- pass 2
__global__ void __launch_bounds__(128)
moe_ffn2_kernel(const unsigned short* __restrict__ hbuf,
                const float* __restrict__ w2,   // [E,H,D] == [KTOT][DIM]
                float* __restrict__ out) {
  __shared__ __align__(16) unsigned short At[128 * LDSS];
  __shared__ __align__(16) unsigned short Bt[64 * LDSS];

  const int tid  = threadIdx.x;
  const int lane = tid & 31;
  const int wid  = tid >> 5;
  const int tM0  = blockIdx.x * 128;
  const int tN0  = blockIdx.y * 64;
  const int m16  = lane & 15;
  const int hi   = lane >> 4;

  const v8f vz = {0.f, 0.f, 0.f, 0.f, 0.f, 0.f, 0.f, 0.f};
  v8f acc[2][4];
#pragma unroll
  for (int mi = 0; mi < 2; ++mi)
#pragma unroll
    for (int s = 0; s < 4; ++s) acc[mi][s] = vz;

  const unsigned short* hrow = hbuf + (size_t)(tM0 + tid) * KTOT;
  u32x4* aout = (u32x4*)&At[tid * LDSS];

  for (int k0 = 0; k0 < KTOT; k0 += 32) {
    const u32x4* src = (const u32x4*)(hrow + k0);
#pragma unroll
    for (int j = 0; j < 4; ++j) aout[j] = src[j];
#pragma unroll
    for (int i = 0; i < 4; ++i) {
      int idx = tid + 128 * i;
      int k = idx >> 4;
      int n = (idx & 15) * 4;
      f32x4 w = *(const f32x4*)(w2 + (size_t)(k0 + k) * DIM + tN0 + n);
      Bt[(n + 0) * LDSS + k] = f2bf(w.x);
      Bt[(n + 1) * LDSS + k] = f2bf(w.y);
      Bt[(n + 2) * LDSS + k] = f2bf(w.z);
      Bt[(n + 3) * LDSS + k] = f2bf(w.w);
    }
    __syncthreads();

    Frag fa0, fa1;
    const u32x4* ar0 = (const u32x4*)&At[(wid * 32 + m16) * LDSS];
    fa0.q[0] = ar0[hi];
    fa0.q[1] = ar0[2 + hi];
    const u32x4* ar1 = (const u32x4*)&At[(wid * 32 + 16 + m16) * LDSS];
    fa1.q[0] = ar1[hi];
    fa1.q[1] = ar1[2 + hi];
#pragma unroll
    for (int s = 0; s < 4; ++s) {
      Frag fb;
      const u32x4* br = (const u32x4*)&Bt[(s * 16 + m16) * LDSS];
      fb.q[0] = br[hi]; fb.q[1] = br[2 + hi];
      acc[0][s] = __builtin_amdgcn_wmma_f32_16x16x32_bf16(
          false, fa0.v, false, fb.v, (short)0, acc[0][s], false, false);
      acc[1][s] = __builtin_amdgcn_wmma_f32_16x16x32_bf16(
          false, fa1.v, false, fb.v, (short)0, acc[1][s], false, false);
    }
    __syncthreads();
  }

#pragma unroll
  for (int mi = 0; mi < 2; ++mi) {
#pragma unroll
    for (int s = 0; s < 4; ++s) {
#pragma unroll
      for (int r = 0; r < 8; ++r) {
        int t = tM0 + wid * 32 + mi * 16 + hi * 8 + r;
        int c = tN0 + s * 16 + m16;
        out[(size_t)t * DIM + c] = acc[mi][s][r];
      }
    }
  }
}

// ---------------------------------------------------------------- launch
extern "C" void kernel_launch(void* const* d_in, const int* in_sizes, int n_in,
                              void* d_out, int out_size, void* d_ws, size_t ws_size,
                              hipStream_t stream) {
  (void)in_sizes; (void)n_in; (void)out_size; (void)ws_size;
  const float* x  = (const float*)d_in[0];
  const float* gw = (const float*)d_in[1];
  const float* w1 = (const float*)d_in[2];
  const float* w2 = (const float*)d_in[3];
  const float* w3 = (const float*)d_in[4];
  float* out = (float*)d_out;

  float* scale = (float*)d_ws;                                   // 32 KB
  unsigned short* hbuf = (unsigned short*)((char*)d_ws + 32768); // 8 MB bf16

  moe_route_kernel<<<dim3(4), dim3(256), 0, stream>>>(x, gw, scale);
  moe_ffn1_kernel<<<dim3(T_TOK / 128, HID / 64, NEXP), dim3(128), 0, stream>>>(
      x, scale, w1, w3, hbuf);
  moe_ffn2_kernel<<<dim3(T_TOK / 128, DIM / 64), dim3(128), 0, stream>>>(
      hbuf, w2, out);
}
